// Field_62706522522369
// MI455X (gfx1250) — compile-verified, matrix operand order checked by reading, not run on hardware
//
#include <hip/hip_runtime.h>
#include <hip/hip_bf16.h>

typedef __attribute__((ext_vector_type(16))) _Float16 v16h;
typedef __attribute__((ext_vector_type(8)))  float    v8f;
typedef __attribute__((ext_vector_type(4)))  int      v4i;

#if defined(__AMDGCN__) && __has_builtin(__builtin_amdgcn_global_load_async_to_lds_b128) && __has_builtin(__builtin_amdgcn_s_wait_asynccnt)
#define USE_ASYNC 1
#else
#define USE_ASYNC 0
#endif

// ---------------- workspace layout (bytes) ----------------
#define WS_BASE1 0          // B*256 f32 (<=4KB)
#define WS_W1R   4096       // 3*256 f32 (xyz rows of W1)
#define WS_W2B   8192       // 65536 f16 swizzled W2  (128KB)
#define WS_W3B   139264     // 32768 f16 swizzled W3  (64KB)

// ---------------- LDS layout (bytes) ----------------
#define OFF_W2   0          // 131072
#define OFF_W3   131072     // 65536
#define OFF_B1   196608     // 256 f32 (base1 for current batch)
#define OFF_W1R  197632     // 768 f32
#define OFF_B2   200704     // 256 f32
#define OFF_B3   201728     // 128 f32
#define OFF_WSV  202240     // 128 f32
#define OFF_WC   202752     // 393 f32 (rounded to 204352)
#define OFF_H2   204352     // 4 waves * 4096 f16 = 32768
#define OFF_FEAT 237120     // 4 waves * 2048 f32 = 32768
#define SMEM_SZ  269888

// base1[b][j] = b1[j] + sum_k global_feat[b][k] * W1[k][j]   (folds layer-1's 256-wide part)
__global__ void prep_base1(const float* __restrict__ gf, const float* __restrict__ W1,
                           const float* __restrict__ b1, float* __restrict__ base1) {
    int b = blockIdx.x, j = threadIdx.x;           // blockDim = 256
    float acc = b1[j];
    for (int k = 0; k < 256; ++k) acc += gf[b * 256 + k] * W1[k * 256 + j];
    base1[b * 256 + j] = acc;
}

// copy W1 rows 256..258 (the xyz rows)
__global__ void prep_w1r(const float* __restrict__ W1, float* __restrict__ w1r) {
    int i = blockIdx.x * blockDim.x + threadIdx.x;
    if (i < 768) w1r[i] = W1[65536 + i];
}

// Pre-swizzle W2/W3 into WMMA B-fragment order:
// blob[((nt*8+ks)*32+lane)*16 + t] = W[K][N], K = ks*32 + (v/4)*16 + hi*8 + (v%4)*2 + s,
// N = nt*16 + (lane%16), v=t/2, s=t%2, hi=lane/16 (mirrors the documented 16-bit A layout).
__global__ void prep_wblob(const float* __restrict__ W2, const float* __restrict__ W3,
                           _Float16* __restrict__ w2b, _Float16* __restrict__ w3b) {
    int id = blockIdx.x * blockDim.x + threadIdx.x;
    const float* W; _Float16* dst; int ncols; int e;
    if (id < 65536)       { e = id;          W = W2; dst = w2b; ncols = 256; }
    else if (id < 98304)  { e = id - 65536;  W = W3; dst = w3b; ncols = 128; }
    else return;
    int t = e & 15, lane = (e >> 4) & 31, ks = (e >> 9) & 7, nt = e >> 12;
    int hi = lane >> 4, lr = lane & 15, v = t >> 1, sb = t & 1;
    int K = ks * 32 + (v >> 2) * 16 + hi * 8 + (v & 3) * 2 + sb;
    int N = nt * 16 + lr;
    dst[e] = (_Float16)W[K * ncols + N];
}

#if USE_ASYNC
__device__ __forceinline__ void async_cp16(const void* g, void* l) {
    // parameter type per hipcc diagnostic: 'int __vector(4) *'
    __builtin_amdgcn_global_load_async_to_lds_b128((v4i*)g, (v4i*)l, 0, 0);
}
#endif

__global__ __launch_bounds__(128, 1)
void field_main(const float* __restrict__ x, const float* __restrict__ ray_dir,
                const unsigned char* __restrict__ mask, const float* __restrict__ offset,
                const float* __restrict__ local_feat,
                const float* __restrict__ b2g, const float* __restrict__ b3g,
                const float* __restrict__ Wsg, const float* __restrict__ bsg,
                const float* __restrict__ Wcg, const float* __restrict__ bcg,
                const float* __restrict__ ws_base1, const float* __restrict__ ws_w1r,
                const _Float16* __restrict__ w2blob, const _Float16* __restrict__ w3blob,
                float* __restrict__ out, int perbatch, int nchunks) {
    extern __shared__ char smem[];
    const int tid = threadIdx.x;

    // ---- stage weights into LDS once per (persistent) block ----
    {
        uint4* s2 = (uint4*)(smem + OFF_W2); const uint4* g2 = (const uint4*)w2blob;
        uint4* s3 = (uint4*)(smem + OFF_W3); const uint4* g3 = (const uint4*)w3blob;
#if USE_ASYNC
        for (int i = tid; i < 8192; i += 128) async_cp16(g2 + i, s2 + i);
        for (int i = tid; i < 4096; i += 128) async_cp16(g3 + i, s3 + i);
#else
        for (int i = tid; i < 8192; i += 128) s2[i] = g2[i];
        for (int i = tid; i < 4096; i += 128) s3[i] = g3[i];
#endif
    }
    float* s_b1  = (float*)(smem + OFF_B1);
    float* s_w1r = (float*)(smem + OFF_W1R);
    float* s_b2  = (float*)(smem + OFF_B2);
    float* s_b3  = (float*)(smem + OFF_B3);
    float* s_ws  = (float*)(smem + OFF_WSV);
    float* s_wc  = (float*)(smem + OFF_WC);
    for (int i = tid; i < 768; i += 128) s_w1r[i] = ws_w1r[i];
    for (int i = tid; i < 256; i += 128) s_b2[i]  = b2g[i];
    for (int i = tid; i < 128; i += 128) s_b3[i]  = b3g[i];
    for (int i = tid; i < 128; i += 128) s_ws[i]  = Wsg[i];
    for (int i = tid; i < 393; i += 128) s_wc[i]  = Wcg[i];
#if USE_ASYNC
    __builtin_amdgcn_s_wait_asynccnt(0);
#endif
    __syncthreads();

    const int wave = tid >> 5, lane = tid & 31, hi = lane >> 4, lr = lane & 15;
    const v16h* w2f = (const v16h*)(smem + OFF_W2);
    const v16h* w3f = (const v16h*)(smem + OFF_W3);
    _Float16* h2buf   = (_Float16*)(smem + OFF_H2) + wave * 4096;
    float*    featbuf = (float*)(smem + OFF_FEAT) + wave * 2048;

    int bb_staged = -1;

    // ---- persistent grid-stride loop over 64-point chunks ----
    for (int chunk = blockIdx.x; chunk < nchunks; chunk += gridDim.x) {
        const int block0 = chunk * 64;
        const int bb = block0 / perbatch;          // block-uniform

        if (bb != bb_staged) {                     // uniform branch: barriers are safe
            __syncthreads();
            for (int i = tid; i < 256; i += 128) s_b1[i] = ws_base1[bb * 256 + i];
            bb_staged = bb;
            __syncthreads();
        }

        // prefetch next chunk's local_feat stream into cache (global_prefetch_b8)
        {
            int nxt = chunk + gridDim.x;
            if (nxt < nchunks) {
                const float* p = local_feat + (size_t)nxt * 64 * 128 + tid * 64;
                __builtin_prefetch(p, 0, 1);
                __builtin_prefetch(p + 32, 0, 1);
            }
        }

        const int tileBase = block0 + wave * 16;
        const int pt = tileBase + lr;              // lane owns matrix row M=lr (both halves)

        // ---- layer 1 (folded): h1 = relu(base1 + xo . W1_xyz) built straight into A fragments
        const float ox = offset[bb * 3 + 0], oy = offset[bb * 3 + 1], oz = offset[bb * 3 + 2];
        const float x0 = x[pt * 3 + 0] - ox, x1 = x[pt * 3 + 1] - oy, x2 = x[pt * 3 + 2] - oz;

        v16h a1[8];
#pragma unroll
        for (int ks = 0; ks < 8; ++ks) {
#pragma unroll
            for (int run = 0; run < 2; ++run) {
#pragma unroll
                for (int i = 0; i < 8; ++i) {
                    int j = ks * 32 + run * 16 + hi * 8 + i;
                    float h = s_b1[j] + x0 * s_w1r[j] + x1 * s_w1r[256 + j] + x2 * s_w1r[512 + j];
                    a1[ks][run * 8 + i] = (_Float16)fmaxf(h, 0.f);
                }
            }
        }

        // ---- layer 2: 16 N-tiles, processed 2 at a time so B-loads overlap WMMAs ----
        for (int nt = 0; nt < 16; nt += 2) {
            v8f acc0 = {}, acc1 = {};
#pragma unroll
            for (int ks = 0; ks < 8; ++ks) {
                v16h bf0 = w2f[((nt + 0) * 8 + ks) * 32 + lane];
                v16h bf1 = w2f[((nt + 1) * 8 + ks) * 32 + lane];
                acc0 = __builtin_amdgcn_wmma_f32_16x16x32_f16(false, a1[ks], false, bf0,
                                                              (short)0, acc0, false, false);
                acc1 = __builtin_amdgcn_wmma_f32_16x16x32_f16(false, a1[ks], false, bf1,
                                                              (short)0, acc1, false, false);
            }
#pragma unroll
            for (int half = 0; half < 2; ++half) {
                const v8f acc = half ? acc1 : acc0;
                const int K = (nt + half) * 16 + lr;   // output feature index (lane-constant)
                const float bias = s_b2[K];
                const int ks2 = K >> 5, o = K & 31;
                const int hi2 = (o & 15) >> 3;
                const int v2  = (o >> 4) * 4 + ((o & 7) >> 1);
                const int t2  = 2 * v2 + (o & 1);
#pragma unroll
                for (int v = 0; v < 8; ++v) {
                    int M = v + 8 * hi;                // D layout: M = vgpr + 8*(lane>=16)
                    float hv = fmaxf(acc[v] + bias, 0.f);
                    h2buf[((hi2 * 16 + M) * 8 + ks2) * 16 + t2] = (_Float16)hv;
                }
            }
        }

        // ---- layer 3: 8 N-tiles, 2 at a time ----
        v16h a2[8];
#pragma unroll
        for (int ks = 0; ks < 8; ++ks)
            a2[ks] = *(const v16h*)(h2buf + (lane * 8 + ks) * 16);

        for (int nt = 0; nt < 8; nt += 2) {
            v8f acc0 = {}, acc1 = {};
#pragma unroll
            for (int ks = 0; ks < 8; ++ks) {
                v16h bf0 = w3f[((nt + 0) * 8 + ks) * 32 + lane];
                v16h bf1 = w3f[((nt + 1) * 8 + ks) * 32 + lane];
                acc0 = __builtin_amdgcn_wmma_f32_16x16x32_f16(false, a2[ks], false, bf0,
                                                              (short)0, acc0, false, false);
                acc1 = __builtin_amdgcn_wmma_f32_16x16x32_f16(false, a2[ks], false, bf1,
                                                              (short)0, acc1, false, false);
            }
#pragma unroll
            for (int half = 0; half < 2; ++half) {
                const v8f acc = half ? acc1 : acc0;
                const int N = (nt + half) * 16 + lr;
                const float b3v = s_b3[N];
#pragma unroll
                for (int v = 0; v < 8; ++v) {
                    int M = v + 8 * hi;
                    featbuf[M * 128 + N] = acc[v] + b3v;
                }
            }
        }

        // ---- epilogue: feat += local_feat; heads. Half-wave splits the 128-dim dots ----
        float s = 0.f, c0 = 0.f, c1 = 0.f, c2 = 0.f;
        const float* lf = local_feat + (size_t)pt * 128;
        const int h0 = hi * 64;
#pragma unroll 4
        for (int h = h0; h < h0 + 64; ++h) {
            float f = featbuf[lr * 128 + h] + lf[h];
            s  += f * s_ws[h];
            c0 += f * s_wc[h * 3 + 0];
            c1 += f * s_wc[h * 3 + 1];
            c2 += f * s_wc[h * 3 + 2];
        }
        s  += __shfl_xor(s, 16, 32);
        c0 += __shfl_xor(c0, 16, 32);
        c1 += __shfl_xor(c1, 16, 32);
        c2 += __shfl_xor(c2, 16, 32);

        if (hi == 0) {
            const int ray = pt >> 5;               // P = 32
            const float r0 = ray_dir[ray * 3 + 0], r1 = ray_dir[ray * 3 + 1], r2 = ray_dir[ray * 3 + 2];
            float cc0 = c0 + r0 * s_wc[384 + 0] + r1 * s_wc[387 + 0] + r2 * s_wc[390 + 0] + bcg[0];
            float cc1 = c1 + r0 * s_wc[384 + 1] + r1 * s_wc[387 + 1] + r2 * s_wc[390 + 1] + bcg[1];
            float cc2 = c2 + r0 * s_wc[384 + 2] + r1 * s_wc[387 + 2] + r2 * s_wc[390 + 2] + bcg[2];
            float z  = s + bsg[0] - 1.f;
            float sp = (z > 20.f) ? z : log1pf(__expf(z));     // softplus
            float m  = mask[pt] ? 1.f : 0.f;
            out[(size_t)pt * 4 + 0] = m * sp;
            out[(size_t)pt * 4 + 1] = m * (1.f / (1.f + __expf(-cc0)));
            out[(size_t)pt * 4 + 2] = m * (1.f / (1.f + __expf(-cc1)));
            out[(size_t)pt * 4 + 3] = m * (1.f / (1.f + __expf(-cc2)));
        }
        // per-wave LDS buffers (h2buf/featbuf) are reused next chunk by the same wave:
        // DS ops are in-order within a wave, so no block barrier is needed here.
    }
}

extern "C" void kernel_launch(void* const* d_in, const int* in_sizes, int n_in,
                              void* d_out, int out_size, void* d_ws, size_t ws_size,
                              hipStream_t stream) {
    const float* x        = (const float*)d_in[0];
    const float* ray_dir  = (const float*)d_in[1];
    const unsigned char* mask = (const unsigned char*)d_in[2];
    const float* gf       = (const float*)d_in[3];
    const float* offset   = (const float*)d_in[4];
    const float* lf       = (const float*)d_in[5];
    const float* W1       = (const float*)d_in[6];
    const float* b1       = (const float*)d_in[7];
    const float* W2       = (const float*)d_in[8];
    const float* b2       = (const float*)d_in[9];
    const float* W3       = (const float*)d_in[10];
    const float* b3       = (const float*)d_in[11];
    const float* Wsv      = (const float*)d_in[12];
    const float* bs       = (const float*)d_in[13];
    const float* Wc       = (const float*)d_in[14];
    const float* bc       = (const float*)d_in[15];
    float* out = (float*)d_out;

    const int npts = in_sizes[2];          // B*T*R*P
    const int B    = in_sizes[4] / 3;      // offset is [B,3]
    const int perb = npts / B;
    const int nchunks = npts / 64;

    char* wsb = (char*)d_ws;
    float*    ws_base1 = (float*)(wsb + WS_BASE1);
    float*    ws_w1r   = (float*)(wsb + WS_W1R);
    _Float16* w2blob   = (_Float16*)(wsb + WS_W2B);
    _Float16* w3blob   = (_Float16*)(wsb + WS_W3B);

    prep_base1<<<B, 256, 0, stream>>>(gf, W1, b1, ws_base1);
    prep_w1r<<<3, 256, 0, stream>>>(W1, ws_w1r);
    prep_wblob<<<(98304 + 255) / 256, 256, 0, stream>>>(W2, W3, w2blob, w3blob);

    int blocks = nchunks < 512 ? nchunks : 512;    // persistent-ish blocks
    field_main<<<blocks, 128, SMEM_SZ, stream>>>(
        x, ray_dir, mask, offset, lf, b2, b3, Wsv, bs, Wc, bc,
        ws_base1, ws_w1r, w2blob, w3blob, out, perb, nchunks);
}
